// CausalSelfAttention_52381421142602
// MI455X (gfx1250) — compile-verified, matrix operand order checked by reading, not run on hardware
//
#include <hip/hip_runtime.h>
#include <hip/hip_bf16.h>

// Causal self-attention forward for MI455X (gfx1250), bf16 WMMA + f32 accum.
// B=4, T=2048, C=1024, H=16, hd=64.

#define DM 1024
#define NH 16
#define HD 64
#define TT 2048

typedef __attribute__((ext_vector_type(16))) __bf16 v16bf;
typedef __attribute__((ext_vector_type(8)))  __bf16 v8bf;
typedef __attribute__((ext_vector_type(8)))  float  v8f;

static __device__ __forceinline__ v16bf pack16(v8bf lo, v8bf hi) {
  return __builtin_shufflevector(lo, hi, 0, 1, 2, 3, 4, 5, 6, 7,
                                         8, 9, 10, 11, 12, 13, 14, 15);
}

// Load one 16x32 (M/N x K) bf16 fragment from a row-major f32 tile.
// Lane layout (A and B fragments are mirrors): row = lane&15,
// K(i) = i + (i>=8 ? 8 : 0) + (lane>=16 ? 8 : 0)  -> two contiguous 8-float runs.
// Vector fptrunc lowers to packed v_cvt_pk_bf16_f32 (results land pre-packed
// in the WMMA operand register layout).
static __device__ __forceinline__ v16bf load_frag(const float* p, int lane, float scale) {
  const int g0 = (lane & 16) ? 8 : 0;
  v8f x0 = *(const v8f*)(p + g0);        // 32B aligned by construction
  v8f x1 = *(const v8f*)(p + g0 + 16);
  x0 *= scale;
  x1 *= scale;
  return pack16(__builtin_convertvector(x0, v8bf),
                __builtin_convertvector(x1, v8bf));
}

// Same fragment gather, but from a bf16 LDS tile with padded row stride.
static __device__ __forceinline__ v16bf lds_frag(const __bf16* row, int g0) {
  return pack16(*(const v8bf*)(row + g0), *(const v8bf*)(row + g0 + 16));
}

#define WMMA_BF16(a, b, c) \
  __builtin_amdgcn_wmma_f32_16x16x32_bf16(false, (a), false, (b), (short)0, (c), false, false)

// ---------------------------------------------------------------------------
// GEMM core: Y[M,N] = X[M,K] * W[N,K]^T. Block: 256 thr = 8 waves -> 128x256
// tile; wave (wm,wn) in 2x4 grid computes 64x64 (4x4 WMMA tiles).
// ---------------------------------------------------------------------------
template <int NMI, int NNI>
static __device__ __forceinline__ void gemm_core(const float* __restrict__ X,
                                                 const float* __restrict__ W,
                                                 int K, int rowA, int rowB, int lane,
                                                 v8f acc[NMI][NNI]) {
#pragma unroll 2
  for (int k0 = 0; k0 < K; k0 += 32) {
    __builtin_prefetch(X + rowA * K + k0 + 128, 0, 3);  // global_prefetch_b8
    v16bf a[NMI], b[NNI];
#pragma unroll
    for (int mi = 0; mi < NMI; ++mi)
      a[mi] = load_frag(X + (rowA + mi * 16) * K + k0, lane, 1.0f);
#pragma unroll
    for (int ni = 0; ni < NNI; ++ni)
      b[ni] = load_frag(W + (rowB + ni * 16) * K + k0, lane, 1.0f);
#pragma unroll
    for (int mi = 0; mi < NMI; ++mi)
#pragma unroll
      for (int ni = 0; ni < NNI; ++ni)
        acc[mi][ni] = WMMA_BF16(a[mi], b[ni], acc[mi][ni]);
  }
}

// ---------------------------------------------------------------------------
// Kernel 1: QKV projection. Writes Q,K as [b,h,t,d], V transposed [b,h,d,t].
// Grid: (3072/256, 8192/128) = (12, 64), block 256.
// ---------------------------------------------------------------------------
__global__ __launch_bounds__(256) void qkv_kernel(const float* __restrict__ X,
                                                  const float* __restrict__ Wqkv,
                                                  const float* __restrict__ bias,
                                                  float* __restrict__ Qb,
                                                  float* __restrict__ Kb,
                                                  float* __restrict__ Vt) {
  const int lane = threadIdx.x & 31, wave = threadIdx.x >> 5;
  const int rlane = lane & 15, half8 = (lane & 16) ? 8 : 0;
  const int wm = wave >> 2, wn = wave & 3;
  const int bm0 = blockIdx.y * 128 + wm * 64;   // M tile base (bt rows)
  const int bn0 = blockIdx.x * 256 + wn * 64;   // N tile base (output cols, 0..3071)
  const int K = DM;

  v8f acc[4][4];
#pragma unroll
  for (int mi = 0; mi < 4; ++mi)
#pragma unroll
    for (int ni = 0; ni < 4; ++ni) acc[mi][ni] = (v8f){};

  gemm_core<4, 4>(X, Wqkv, K, bm0 + rlane, bn0 + rlane, lane, acc);

#pragma unroll
  for (int mi = 0; mi < 4; ++mi) {
#pragma unroll
    for (int ni = 0; ni < 4; ++ni) {
      const int col = bn0 + ni * 16 + rlane;
      const float bv = bias[col];
      const int which = col >> 10;          // 0=q, 1=k, 2=v (uniform per block)
      const int hh = (col >> 6) & 15;
      const int d = col & 63;
#pragma unroll
      for (int e = 0; e < 8; ++e) {
        const int row = bm0 + mi * 16 + e + half8;  // bt
        const int bb = row >> 11;
        const int t = row & 2047;
        const float val = acc[mi][ni][e] + bv;
        if (which == 0)      Qb[(((bb << 4) + hh) * TT + t) * HD + d] = val;
        else if (which == 1) Kb[(((bb << 4) + hh) * TT + t) * HD + d] = val;
        else                 Vt[(((bb << 4) + hh) * HD + d) * TT + t] = val;
      }
    }
  }
}

// ---------------------------------------------------------------------------
// Kernel 2: causal flash attention. Block = 8 waves sharing (b,h) and a
// 128-row q stripe; each 32-row K/V tile is staged into LDS once per block
// (converted to bf16 during staging), then all waves gather WMMA fragments
// from LDS. S^T = K*Q^T keeps softmax stats per-lane; P feeds PV directly as
// the B operand; O accumulated as O^T = V^T * P^T.
// Grid: (T/128, H, B) = (16, 16, 4), block 256.
// ---------------------------------------------------------------------------
#define KLD 72   // padded bf16 row stride for K tile (32 x 64)
#define VLD 40   // padded bf16 row stride for V^T tile (64 x 32)

__global__ __launch_bounds__(256) void attn_kernel(const float* __restrict__ Qb,
                                                   const float* __restrict__ Kb,
                                                   const float* __restrict__ Vt,
                                                   float* __restrict__ Ob) {
  __shared__ __bf16 Kls[32 * KLD];
  __shared__ __bf16 Vls[64 * VLD];

  const int tid = threadIdx.x;
  const int lane = tid & 31, wave = tid >> 5;
  const int rlane = lane & 15, half8 = (lane & 16) ? 8 : 0;
  const int b = blockIdx.z, h = blockIdx.y;
  const int qstripe = blockIdx.x * 128;
  const int qbase = qstripe + wave * 16;

  const int bh = b * NH + h;
  const float* Qp = Qb + bh * (TT * HD);
  const float* Kp = Kb + bh * (TT * HD);
  const float* Vp = Vt + bh * (HD * TT);

  // Q^T as B operand (lane = q column), pre-scaled by 1/sqrt(hd)=0.125 (exact)
  v16bf qf0 = load_frag(Qp + (qbase + rlane) * HD, lane, 0.125f);
  v16bf qf1 = load_frag(Qp + (qbase + rlane) * HD + 32, lane, 0.125f);

  v8f od[4] = {};                 // O^T: rows d (4 x 16), cols q
  float m = -3.0e38f, l = 0.0f;   // per-q running max / sum (per lane)
  const int qp = qbase + rlane;

  // cooperative staging indices (256 threads)
  const int kr = tid >> 3, kc = (tid & 7) * 8;   // K tile: 32 rows x 64 d
  const int vr = tid >> 2, vc = (tid & 3) * 8;   // V^T tile: 64 d x 32 kv

  const int kvend = qstripe + 128;               // uniform loop bound per block
  for (int kv0 = 0; kv0 < kvend; kv0 += 32) {
    __syncthreads();  // previous iteration's LDS reads complete
    {
      v8f kx = *(const v8f*)(Kp + (kv0 + kr) * HD + kc);
      *(v8bf*)&Kls[kr * KLD + kc] = __builtin_convertvector(kx, v8bf);
      v8f vx = *(const v8f*)(Vp + vr * TT + kv0 + vc);
      *(v8bf*)&Vls[vr * VLD + vc] = __builtin_convertvector(vx, v8bf);
    }
    __syncthreads();

    if (kv0 < qbase + 16) {       // this wave's causal range
      // K rows as A operand (lane = kv row), gathered from LDS
      v16bf ka00 = lds_frag(&Kls[rlane * KLD], half8);
      v16bf ka01 = lds_frag(&Kls[rlane * KLD + 32], half8);
      v16bf ka10 = lds_frag(&Kls[(16 + rlane) * KLD], half8);
      v16bf ka11 = lds_frag(&Kls[(16 + rlane) * KLD + 32], half8);

      v8f s0 = (v8f){};
      s0 = WMMA_BF16(ka00, qf0, s0);
      s0 = WMMA_BF16(ka01, qf1, s0);
      v8f s1 = (v8f){};
      s1 = WMMA_BF16(ka10, qf0, s1);
      s1 = WMMA_BF16(ka11, qf1, s1);

      if (kv0 + 31 > qbase) {  // causal mask (kvpos > qpos -> -inf)
#pragma unroll
        for (int e = 0; e < 8; ++e) {
          const int kvp = kv0 + e + half8;
          if (kvp > qp)      s0[e] = -__builtin_inff();
          if (kvp + 16 > qp) s1[e] = -__builtin_inff();
        }
      }

      // online softmax over kv axis (rows of S^T): in-lane max + partner-half
      float tm = -3.0e38f;
#pragma unroll
      for (int e = 0; e < 8; ++e) tm = fmaxf(tm, fmaxf(s0[e], s1[e]));
      tm = fmaxf(tm, __shfl_xor(tm, 16, 32));
      const float mnew = fmaxf(m, tm);
      const float corr = __expf(m - mnew);

      float tsum = 0.0f;
      v8f p0v, p1v;
#pragma unroll
      for (int e = 0; e < 8; ++e) {
        p0v[e] = __expf(s0[e] - mnew);
        p1v[e] = __expf(s1[e] - mnew);
        tsum += p0v[e] + p1v[e];
      }
      // P as B operand: elems 0-7 <- s0 tile, 8-15 <- s1 tile (layouts coincide)
      v16bf pf = pack16(__builtin_convertvector(p0v, v8bf),
                        __builtin_convertvector(p1v, v8bf));
      tsum += __shfl_xor(tsum, 16, 32);
      l = l * corr + tsum;
      m = mnew;
#pragma unroll
      for (int t = 0; t < 4; ++t) od[t] *= corr;

      // O^T += V^T * P^T (V^T rows are d thanks to transposed V layout)
#pragma unroll
      for (int ds = 0; ds < 4; ++ds) {
        v16bf va = lds_frag(&Vls[(ds * 16 + rlane) * VLD], half8);
        od[ds] = WMMA_BF16(va, pf, od[ds]);
      }
    }
  }

  const float inv = 1.0f / l;
  float* Op = Ob + (b * TT + qbase + rlane) * DM + h * HD;
#pragma unroll
  for (int ds = 0; ds < 4; ++ds)
#pragma unroll
    for (int e = 0; e < 8; ++e)
      Op[ds * 16 + e + half8] = od[ds][e] * inv;
}

// ---------------------------------------------------------------------------
// Kernel 3: output projection. Grid: (1024/256, 8192/128) = (4, 64), block 256.
// ---------------------------------------------------------------------------
__global__ __launch_bounds__(256) void proj_kernel(const float* __restrict__ Xo,
                                                   const float* __restrict__ Wo,
                                                   const float* __restrict__ bias,
                                                   float* __restrict__ Y) {
  const int lane = threadIdx.x & 31, wave = threadIdx.x >> 5;
  const int rlane = lane & 15, half8 = (lane & 16) ? 8 : 0;
  const int wm = wave >> 2, wn = wave & 3;
  const int bm0 = blockIdx.y * 128 + wm * 64;
  const int bn0 = blockIdx.x * 256 + wn * 64;

  v8f acc[4][4];
#pragma unroll
  for (int mi = 0; mi < 4; ++mi)
#pragma unroll
    for (int ni = 0; ni < 4; ++ni) acc[mi][ni] = (v8f){};

  gemm_core<4, 4>(Xo, Wo, DM, bm0 + rlane, bn0 + rlane, lane, acc);

#pragma unroll
  for (int mi = 0; mi < 4; ++mi) {
#pragma unroll
    for (int ni = 0; ni < 4; ++ni) {
      const int col = bn0 + ni * 16 + rlane;
      const float bv = bias[col];
#pragma unroll
      for (int e = 0; e < 8; ++e) {
        const int row = bm0 + mi * 16 + e + half8;
        Y[row * DM + col] = acc[mi][ni][e] + bv;
      }
    }
  }
}

// ---------------------------------------------------------------------------
extern "C" void kernel_launch(void* const* d_in, const int* in_sizes, int n_in,
                              void* d_out, int out_size, void* d_ws, size_t ws_size,
                              hipStream_t stream) {
  const float* x      = (const float*)d_in[0];
  const float* qkv_w  = (const float*)d_in[1];
  const float* qkv_b  = (const float*)d_in[2];
  const float* out_w  = (const float*)d_in[3];
  const float* out_b  = (const float*)d_in[4];
  float* out = (float*)d_out;

  float* ws = (float*)d_ws;
  const size_t n = (size_t)4 * NH * TT * HD;  // 8,388,608 floats = 32 MB
  float* Qb = ws;
  float* Kb = ws + n;
  float* Vt = ws + 2 * n;
  float* Ob = ws + 3 * n;

  qkv_kernel<<<dim3(12, 64), 256, 0, stream>>>(x, qkv_w, qkv_b, Qb, Kb, Vt);
  attn_kernel<<<dim3(16, 16, 4), 256, 0, stream>>>(Qb, Kb, Vt, Ob);
  proj_kernel<<<dim3(4, 64), 256, 0, stream>>>(Ob, out_w, out_b, out);
}
